// HistEncoder_41154376630383
// MI455X (gfx1250) — matrix operand order repaired and checked by reading, not compiled, not run on hardware
//
#include <hip/hip_runtime.h>
#include <hip/hip_bf16.h>
#include <cstdint>
#include <cstddef>

// ---------------------------------------------------------------------------
// Model dimensions (match reference)
// ---------------------------------------------------------------------------
#define BB     4096
#define NN     32
#define TT     16
#define KK     6
#define DD     256
#define FFD    1024
#define HH     512
#define NHEADS 8
#define RHEADS 4
#define MEMROW (TT + KK)          // 22
#define ROWS   (BB * TT)          // 65536

typedef __attribute__((ext_vector_type(16))) _Float16 v16h;
typedef __attribute__((ext_vector_type(8)))  float    v8f;
typedef __attribute__((ext_vector_type(4)))  float    f32x4;
typedef int v4i32 __attribute__((vector_size(16)));   // matches builtin V4i

union Frag16 { v16h h; f32x4 f[2]; };

#define DEV static __device__ __forceinline__

#ifndef __has_builtin
#define __has_builtin(x) 0
#endif

// CDNA5 async global->LDS copy (ASYNCcnt-tracked); guarded per-toolchain.
#if defined(__HIP_DEVICE_COMPILE__) && __has_builtin(__builtin_amdgcn_global_load_async_to_lds_b128)
#define HAVE_ASYNC_CP 1
#else
#define HAVE_ASYNC_CP 0
#endif
#if defined(__HIP_DEVICE_COMPILE__) && __has_builtin(__builtin_amdgcn_s_wait_asynccnt)
#define HAVE_WAIT_ASYNC 1
#else
#define HAVE_WAIT_ASYNC 0
#endif

DEV void cp16_g2l(_Float16* dst_lds, const _Float16* src_glb) {
#if HAVE_ASYNC_CP
  __builtin_amdgcn_global_load_async_to_lds_b128(
      (__attribute__((address_space(1))) v4i32*)(src_glb),
      (__attribute__((address_space(3))) v4i32*)(dst_lds), 0, 0);
#else
  *(uint4*)dst_lds = *(const uint4*)src_glb;
#endif
}

DEV void wait_async_3() {
#if HAVE_ASYNC_CP
#if HAVE_WAIT_ASYNC
  __builtin_amdgcn_s_wait_asynccnt(3);
#else
  asm volatile("s_wait_asynccnt 0x3" ::: "memory");
#endif
#endif
}

DEV void wait_async_0() {
#if HAVE_ASYNC_CP
#if HAVE_WAIT_ASYNC
  __builtin_amdgcn_s_wait_asynccnt(0);
#else
  asm volatile("s_wait_asynccnt 0x0" ::: "memory");
#endif
#endif
}

DEV float clampf(float v, float lo, float hi) { return fminf(fmaxf(v, lo), hi); }
DEV float gelu_erf(float x)  { return 0.5f * x * (1.0f + erff(x * 0.70710678118654752f)); }
DEV float gelu_tanh_(float x){ float x3 = x*x*x; return 0.5f * x * (1.0f + tanhf(0.79788456080286536f * (x + 0.044715f * x3))); }
DEV float silu_(float x)     { return x / (1.0f + expf(-x)); }

template<int ACT> DEV float apply_act(float v) {
  if (ACT == 1) return gelu_erf(v);
  if (ACT == 2) return gelu_tanh_(v);
  if (ACT == 3) return silu_(v);
  return v;
}

// ---------------------------------------------------------------------------
// Generic WMMA GEMM:  C = act(A(f16,MxK) @ W + bias) [* rowScale]
// W supplied PRE-TRANSPOSED as Wt (N x K row-major) so both tiles stage as
// contiguous 16B chunks via async global->LDS copies; double-buffered LDS.
// Block: 256 threads (8 wave32), tile 128x64; wave tile 32x32 (2x2 wmma).
// ---------------------------------------------------------------------------
template<int ACT>
__global__ __launch_bounds__(256)
void wmma_gemm(const _Float16* __restrict__ A, const _Float16* __restrict__ Wt,
               const float* __restrict__ bias,
               float* __restrict__ C32, _Float16* __restrict__ C16,
               const float* __restrict__ rowScale,
               int M, int N, int K, int rdiv, int rmul, int radd)
{
  __shared__ _Float16 As[2][128 * 40];   // [m][k], row stride 40 halves
  __shared__ _Float16 Bs[2][64 * 40];    // [n][k], row stride 40 halves

  const int tid  = threadIdx.x;
  const int wave = tid >> 5;
  const int lane = tid & 31;
  const int wm   = wave >> 1;            // 0..3
  const int wn   = wave & 1;             // 0..1
  const int m16  = lane & 15;
  const int kh   = lane >> 4;            // half-wave id
  const int blockM = blockIdx.x * 128;
  const int blockN = blockIdx.y * 64;

  // per-thread staging coordinates
  const int ar0 = tid >> 2;              // A rows for chunk 0 (seg = tid)
  const int ac0 = (tid & 3) << 3;
  const int ar1 = (tid + 256) >> 2;      // A rows for chunk 1
  const int ac1 = ac0;
  const int br  = tid >> 2;              // B row (n)
  const int bc  = (tid & 3) << 3;

  v8f acc[2][2] = {};

  auto stage = [&](int k0, int buf) {
    cp16_g2l(&As[buf][ar0 * 40 + ac0], A  + (size_t)(blockM + ar0) * K + k0 + ac0);
    cp16_g2l(&As[buf][ar1 * 40 + ac1], A  + (size_t)(blockM + ar1) * K + k0 + ac1);
    cp16_g2l(&Bs[buf][br  * 40 + bc ], Wt + (size_t)(blockN + br ) * K + k0 + bc);
    if (k0 + 64 < K) {                    // L2 prefetch two steps ahead
      __builtin_prefetch(A  + (size_t)(blockM + ar0) * K + k0 + 64 + ac0, 0, 1);
      __builtin_prefetch(Wt + (size_t)(blockN + br ) * K + k0 + 64 + bc, 0, 1);
    }
  };

  const int nK = K >> 5;
  stage(0, 0);
  for (int ks = 0; ks < nK; ++ks) {
    const int cur = ks & 1;
    const bool more = (ks + 1 < nK);
    if (more) stage((ks + 1) << 5, cur ^ 1);
    if (more) wait_async_3(); else wait_async_0();
    __syncthreads();

    Frag16 afr[2], bfr[2];
    #pragma unroll
    for (int mt = 0; mt < 2; ++mt) {
      int row = wm * 32 + mt * 16 + m16;
      afr[mt].f[0] = *(const f32x4*)&As[cur][row * 40 + kh * 8];
      afr[mt].f[1] = *(const f32x4*)&As[cur][row * 40 + 16 + kh * 8];
    }
    #pragma unroll
    for (int nt = 0; nt < 2; ++nt) {
      int col = wn * 32 + nt * 16 + m16;
      bfr[nt].f[0] = *(const f32x4*)&Bs[cur][col * 40 + kh * 8];
      bfr[nt].f[1] = *(const f32x4*)&Bs[cur][col * 40 + 16 + kh * 8];
    }
    #pragma unroll
    for (int mt = 0; mt < 2; ++mt)
      #pragma unroll
      for (int nt = 0; nt < 2; ++nt)
        acc[mt][nt] = __builtin_amdgcn_wmma_f32_16x16x32_f16(
            false, afr[mt].h, false, bfr[nt].h, (short)0, acc[mt][nt], false, false);
    __syncthreads();
  }

  // ---- epilogue
  #pragma unroll
  for (int mt = 0; mt < 2; ++mt) {
    #pragma unroll
    for (int nt = 0; nt < 2; ++nt) {
      int n = blockN + wn * 32 + nt * 16 + m16;
      #pragma unroll
      for (int r = 0; r < 8; ++r) {
        int m = blockM + wm * 32 + mt * 16 + r + kh * 8;
        float v = acc[mt][nt][r];
        if (bias) v += bias[n];
        v = apply_act<ACT>(v);
        if (rowScale) v *= rowScale[m];
        if (C16) C16[(size_t)m * N + n] = (_Float16)v;
        if (C32) {
          int orow = m;
          if (rdiv > 0) orow = (m / rdiv) * rmul + radd + (m % rdiv);
          C32[(size_t)orow * N + n] = v;
        }
      }
    }
  }
}

// ---------------------------------------------------------------------------
// f32 (K x N) -> f16 transposed (N x K)
// ---------------------------------------------------------------------------
__global__ void f2h_t_kernel(const float* __restrict__ s, _Float16* __restrict__ d,
                             int K, int N)
{
  int i = blockIdx.x * 256 + threadIdx.x;
  if (i < K * N) {
    int k = i / N, n = i % N;
    d[(size_t)n * K + k] = (_Float16)s[i];
  }
}

// ---------------------------------------------------------------------------
// Embedding + sinusoidal positional encoding; writes x, pos, x16, qk16
// ---------------------------------------------------------------------------
__global__ __launch_bounds__(256)
void embed_pos_kernel(const float* __restrict__ esn,
                      const float* __restrict__ emb_w, const float* __restrict__ emb_b,
                      float* __restrict__ x, float* __restrict__ pos,
                      _Float16* __restrict__ x16, _Float16* __restrict__ qk16)
{
  size_t idx = (size_t)blockIdx.x * 256 + threadIdx.x;   // ROWS*DD elements
  int row = (int)(idx >> 8);
  int d   = (int)(idx & 255);
  const float* s = esn + (size_t)row * 4;
  float xv = emb_b[d];
  #pragma unroll
  for (int j = 0; j < 4; ++j) xv += s[j] * emb_w[j * DD + d];

  float e = (d < 128) ? s[1] : s[0];
  int i   = d & 127;
  float ex   = (float)((i >> 1) * 2) / 128.0f;
  float dimt = powf(10000.0f, ex);
  float ph   = e * 6.2831853071795864769f / dimt;
  float pv   = (i & 1) ? cosf(ph) : sinf(ph);

  x[idx]   = xv;
  pos[idx] = pv;
  x16[idx]  = (_Float16)xv;
  qk16[idx] = (_Float16)(xv + pv);
}

// ---------------------------------------------------------------------------
// LayerNorm (optionally residual add), optional f32/f16 outputs, optional
// qk16 = f16(out + pos). Block per row, ncols = 256 or 512.
// ---------------------------------------------------------------------------
__global__ __launch_bounds__(256)
void ln_kernel(const float* __restrict__ in, const float* __restrict__ add,
               const float* __restrict__ g, const float* __restrict__ bia,
               float* __restrict__ out32, _Float16* __restrict__ out16,
               const float* __restrict__ pos, _Float16* __restrict__ qk16, int ncols)
{
  __shared__ float red[256];
  int row = blockIdx.x;
  int tid = threadIdx.x;
  int per = ncols >> 8;                 // 1 or 2
  size_t base = (size_t)row * ncols;

  float v[2];
  float s = 0.0f;
  for (int i = 0; i < per; ++i) {
    int c = tid + (i << 8);
    v[i] = in[base + c];
    if (add) v[i] += add[base + c];
    s += v[i];
  }
  red[tid] = s; __syncthreads();
  for (int st = 128; st > 0; st >>= 1) { if (tid < st) red[tid] += red[tid + st]; __syncthreads(); }
  float mean = red[0] / (float)ncols;
  __syncthreads();

  float s2 = 0.0f;
  for (int i = 0; i < per; ++i) { float dd = v[i] - mean; s2 += dd * dd; }
  red[tid] = s2; __syncthreads();
  for (int st = 128; st > 0; st >>= 1) { if (tid < st) red[tid] += red[tid + st]; __syncthreads(); }
  float var  = red[0] / (float)ncols;
  float rstd = rsqrtf(var + 1e-5f);

  for (int i = 0; i < per; ++i) {
    int c = tid + (i << 8);
    float y = (v[i] - mean) * rstd * g[c] + bia[c];
    if (out32) out32[base + c] = y;
    if (out16) out16[base + c] = (_Float16)y;
    if (qk16)  qk16[base + c]  = (_Float16)(y + pos[base + c]);
  }
}

// ---------------------------------------------------------------------------
// Encoder self-attention (T=16, 8 heads, dh=32). Block per batch, 128 thr.
// ---------------------------------------------------------------------------
__global__ __launch_bounds__(128)
void attn_enc_kernel(const float* __restrict__ Q, const float* __restrict__ Kb,
                     const float* __restrict__ Vb,
                     float* __restrict__ attn32, _Float16* __restrict__ attn16)
{
  int b  = blockIdx.x;
  int t  = threadIdx.x;
  int h  = t >> 4;
  int qi = t & 15;
  size_t qrow = ((size_t)b * TT + qi) * DD + h * 32;

  float q[32];
  #pragma unroll
  for (int j = 0; j < 32; ++j) q[j] = Q[qrow + j];

  float s[16];
  float mx = -1e30f;
  for (int k = 0; k < 16; ++k) {
    size_t kr = ((size_t)b * TT + k) * DD + h * 32;
    float a = 0.0f;
    #pragma unroll
    for (int j = 0; j < 32; ++j) a += q[j] * Kb[kr + j];
    a *= 0.17677669529663687f;          // 1/sqrt(32)
    s[k] = a;
    mx = fmaxf(mx, a);
  }
  float sum = 0.0f;
  for (int k = 0; k < 16; ++k) { s[k] = expf(s[k] - mx); sum += s[k]; }
  float inv = 1.0f / sum;

  float o[32] = {};
  for (int k = 0; k < 16; ++k) {
    float w = s[k] * inv;
    size_t vr = ((size_t)b * TT + k) * DD + h * 32;
    #pragma unroll
    for (int j = 0; j < 32; ++j) o[j] += w * Vb[vr + j];
  }
  #pragma unroll
  for (int j = 0; j < 32; ++j) {
    attn32[qrow + j] = o[j];
    attn16[qrow + j] = (_Float16)o[j];
  }
}

// ---------------------------------------------------------------------------
// Lane-meta embedding + ego_tok + eh LayerNorm -> eh_in16. Block per row.
// ---------------------------------------------------------------------------
__global__ __launch_bounds__(256)
void meta_kernel(const float* __restrict__ x, const float* __restrict__ ego_lane,
                 const float* __restrict__ lane_tab,
                 const float* __restrict__ mg, const float* __restrict__ mb,
                 const float* __restrict__ mw, const float* __restrict__ mbias,
                 const float* __restrict__ ehg, const float* __restrict__ ehb,
                 _Float16* __restrict__ eh16)
{
  __shared__ float red[256];
  int row = blockIdx.x;
  int d   = threadIdx.x;

  float lv  = ego_lane[row];
  int idx   = (int)roundf(lv) - 1;
  idx = idx < 0 ? 0 : (idx > 7 ? 7 : idx);

  float l[8]; float m = 0.0f;
  #pragma unroll
  for (int j = 0; j < 8; ++j) { l[j] = lane_tab[idx * 8 + j]; m += l[j]; }
  m *= 0.125f;
  float var = 0.0f;
  #pragma unroll
  for (int j = 0; j < 8; ++j) { float dd = l[j] - m; var += dd * dd; }
  var *= 0.125f;
  float rstd = rsqrtf(var + 1e-5f);

  float acc = mbias[d];
  #pragma unroll
  for (int j = 0; j < 8; ++j) acc += ((l[j] - m) * rstd * mg[j] + mb[j]) * mw[j * DD + d];
  float tok = x[(size_t)row * DD + d] + gelu_tanh_(acc);

  red[d] = tok; __syncthreads();
  for (int st = 128; st > 0; st >>= 1) { if (d < st) red[d] += red[d + st]; __syncthreads(); }
  float mean = red[0] / (float)DD;
  __syncthreads();
  float dd2 = tok - mean;
  red[d] = dd2 * dd2; __syncthreads();
  for (int st = 128; st > 0; st >>= 1) { if (d < st) red[d] += red[d + st]; __syncthreads(); }
  float rstd2 = rsqrtf(red[0] / (float)DD + 1e-5f);

  eh16[(size_t)row * DD + d] = (_Float16)((tok - mean) * rstd2 * ehg[d] + ehb[d]);
}

// ---------------------------------------------------------------------------
// Mask scan (generic cumsum over B*N mask)
// ---------------------------------------------------------------------------
__global__ void cnt_kernel(const unsigned char* __restrict__ mask, int* __restrict__ cnt)
{
  int b = blockIdx.x * 256 + threadIdx.x;
  if (b < BB) {
    int s = 0;
    for (int i = 0; i < NN; ++i) s += mask[(size_t)b * NN + i] ? 1 : 0;
    cnt[b] = s;
  }
}

__global__ __launch_bounds__(256)
void scan_kernel(const int* __restrict__ cnt, int* __restrict__ base)
{
  __shared__ int red[256];
  int t = threadIdx.x;
  int local[16]; int s = 0;
  #pragma unroll
  for (int i = 0; i < 16; ++i) { local[i] = cnt[t * 16 + i]; s += local[i]; }
  red[t] = s; __syncthreads();
  for (int off = 1; off < 256; off <<= 1) {
    int v = (t >= off) ? red[t - off] : 0;
    __syncthreads();
    red[t] += v;
    __syncthreads();
  }
  int run = red[t] - s;                 // exclusive prefix
  #pragma unroll
  for (int i = 0; i < 16; ++i) { base[t * 16 + i] = run; run += local[i]; }
}

// ---------------------------------------------------------------------------
// Neighbor scoring + top-K selection + rel features + mem_mask. 1 wave/batch.
// ---------------------------------------------------------------------------
__global__ __launch_bounds__(32)
void select_kernel(const unsigned char* __restrict__ mask, const int* __restrict__ base,
                   const float* __restrict__ nbr_raw, const float* __restrict__ nbr_lane,
                   const float* __restrict__ nbr_dist,
                   const float* __restrict__ ego_raw, const float* __restrict__ ego_lane,
                   float* __restrict__ feat, float* __restrict__ selval,
                   float* __restrict__ memmask)
{
  __shared__ float sx[32], sy[32], sv2[32], sa[32], sl[32], sd[32], sscore[32];
  __shared__ int   smask[32], sclose[32], selIdx[8];
  __shared__ float selScore[8];

  int b = blockIdx.x;
  int n = threadIdx.x;

  int mr = mask[(size_t)b * NN + n] ? 1 : 0;
  smask[n] = mr;
  __syncthreads();
  int intra = 0;
  for (int i = 0; i < n; ++i) intra += smask[i];
  int pos = base[b] + intra;

  float nx = 0, ny = 0, nv = 0, na = 0, nl = 0, nd = 0;
  if (mr) {
    const float* p = nbr_raw + ((size_t)pos * TT + (TT - 1)) * 4;
    nx = p[0]; ny = p[1]; nv = p[2]; na = p[3];
    nl = nbr_lane[(size_t)pos * TT + (TT - 1)];
    nd = nbr_dist[(size_t)pos * TT + (TT - 1)];
  }
  sx[n] = nx; sy[n] = ny; sv2[n] = nv; sa[n] = na; sl[n] = nl; sd[n] = nd;

  const float* eg = ego_raw + (size_t)b * TT * 4 + (TT - 1) * 4;
  float ex = eg[0], ey = eg[1], ev = eg[2], ea = eg[3];
  float el = ego_lane[(size_t)b * TT + (TT - 1)];

  float ld    = nl - el;
  float same  = (fabsf(ld) < 0.5f) ? 1.0f : 0.0f;
  float adj   = (fabsf(fabsf(ld) - 1.0f) < 0.5f) ? 1.0f : 0.0f;
  float dx    = fabsf(nx - ex);
  float dy    = fabsf(ny - ey);
  float closing = fmaxf(ev - nv, 0.0f);
  float sc = 1.25f / (dy + 1.0f) + 0.75f / (nd + 1.0f)
           + 0.25f * clampf(closing * 0.1f, 0.0f, 2.0f)
           + 0.2f * same + 0.1f * adj + 0.15f / (dx + 1.0f);

  int close = (nd <= 120.0f) ? 1 : 0;
  sclose[n] = (mr && close) ? 1 : 0;
  __syncthreads();
  int hasClose = 0;
  for (int i = 0; i < 32; ++i) hasClose |= sclose[i];
  int avail = mr && (!hasClose || close);
  sscore[n] = avail ? sc : -__builtin_inff();
  __syncthreads();

  if (n == 0) {
    for (int k = 0; k < KK; ++k) {
      float best = -__builtin_inff(); int bi = 0;
      for (int i = 0; i < 32; ++i) if (sscore[i] > best) { best = sscore[i]; bi = i; }
      selIdx[k] = bi; selScore[k] = best;
      sscore[bi] = -__builtin_inff();
    }
  }
  __syncthreads();

  if (n < TT) memmask[(size_t)b * MEMROW + n] = 0.0f;
  if (n < KK) {
    int k = n;
    int i = selIdx[k];
    float valid = (selScore[k] > -1e30f) ? 1.0f : 0.0f;
    float gx = sx[i], gy = sy[i], gv = sv2[i], ga = sa[i], gl = sl[i], gd = sd[i];
    float ldd = gl - el;
    float* fp = feat + ((size_t)b * KK + k) * 9;
    fp[0] = clampf((gx - ex) / 20.0f, -10.0f, 10.0f);
    fp[1] = clampf((gy - ey) / 80.0f, -10.0f, 10.0f);
    fp[2] = clampf((gv - ev) / 15.0f, -10.0f, 10.0f);
    fp[3] = clampf((ga - ea) / 5.0f,  -10.0f, 10.0f);
    fp[4] = clampf(gd / 120.0f, 0.0f, 2.0f);
    fp[5] = clampf(ldd * 0.5f, -2.0f, 2.0f);
    fp[6] = (fabsf(ldd) < 0.5f) ? 1.0f : 0.0f;
    fp[7] = (fabsf(fabsf(ldd) - 1.0f) < 0.5f) ? 1.0f : 0.0f;
    fp[8] = clampf((ev - gv) * 0.1f, -2.0f, 2.0f);
    selval[(size_t)b * KK + k] = valid;
    memmask[(size_t)b * MEMROW + TT + k] = valid ? 0.0f : 1.0f;
  }
}

// ---------------------------------------------------------------------------
// rel MLP stage 1: s16 = silu(LN9(feat) @ rel_w1 + rel_b1). Block per row.
// ---------------------------------------------------------------------------
__global__ __launch_bounds__(256)
void rel_mlp1_kernel(const float* __restrict__ feat,
                     const float* __restrict__ g9, const float* __restrict__ b9,
                     const float* __restrict__ w1, const float* __restrict__ b1,
                     _Float16* __restrict__ s16)
{
  int row = blockIdx.x;
  int t   = threadIdx.x;
  const float* f = feat + (size_t)row * 9;
  float lf[9]; float m = 0.0f;
  #pragma unroll
  for (int j = 0; j < 9; ++j) { lf[j] = f[j]; m += lf[j]; }
  m /= 9.0f;
  float var = 0.0f;
  #pragma unroll
  for (int j = 0; j < 9; ++j) { float dd = lf[j] - m; var += dd * dd; }
  var /= 9.0f;
  float rstd = rsqrtf(var + 1e-5f);
  float h[9];
  #pragma unroll
  for (int j = 0; j < 9; ++j) h[j] = (lf[j] - m) * rstd * g9[j] + b9[j];

  #pragma unroll
  for (int rep = 0; rep < 2; ++rep) {
    int nn = t + (rep << 8);
    float a = b1[nn];
    #pragma unroll
    for (int j = 0; j < 9; ++j) a += h[j] * w1[j * HH + nn];
    s16[(size_t)row * HH + nn] = (_Float16)silu_(a);
  }
}

// ---------------------------------------------------------------------------
// Intent query projection: Qc[q] = cls_q @ wq + bq (2x512, shared by batches)
// ---------------------------------------------------------------------------
__global__ __launch_bounds__(512)
void qc_kernel(const float* __restrict__ lat_cls, const float* __restrict__ lon_cls,
               const float* __restrict__ wq, const float* __restrict__ bq,
               float* __restrict__ Qc)
{
  int q = blockIdx.x;
  int n = threadIdx.x;
  const float* c = q ? lon_cls : lat_cls;
  float a = bq[n];
  for (int d = 0; d < HH; ++d) a += c[d] * wq[d * HH + n];
  Qc[q * HH + n] = a;
}

// ---------------------------------------------------------------------------
// Intent attention: 2 queries x 4 heads x 22 keys, dh=128. Block per batch.
// ---------------------------------------------------------------------------
__global__ __launch_bounds__(256)
void intent_attn_kernel(const float* __restrict__ Qc, const float* __restrict__ Ki,
                        const float* __restrict__ Vi, const float* __restrict__ memmask,
                        _Float16* __restrict__ att16)
{
  __shared__ float sS[8 * MEMROW];
  int b = blockIdx.x;
  int t = threadIdx.x;

  if (t < 8 * MEMROW) {
    int g = t / MEMROW, k = t % MEMROW;
    int q = g >> 2, h = g & 3;
    const float* kp = Ki + ((size_t)b * MEMROW + k) * HH + h * 128;
    const float* qp = Qc + q * HH + h * 128;
    float a = 0.0f;
    for (int j = 0; j < 128; ++j) a += qp[j] * kp[j];
    a *= 0.08838834764831845f;          // 1/sqrt(128)
    if (memmask[(size_t)b * MEMROW + k] != 0.0f) a = -1e9f;
    sS[g * MEMROW + k] = a;
  }
  __syncthreads();

  for (int i = 0; i < 4; ++i) {
    int idx = t + (i << 8);             // 0..1023
    int g = idx >> 7, d = idx & 127;
    int q = g >> 2, h = g & 3;
    float mx = -1e30f;
    for (int k = 0; k < MEMROW; ++k) mx = fmaxf(mx, sS[g * MEMROW + k]);
    float w[MEMROW]; float sum = 0.0f;
    for (int k = 0; k < MEMROW; ++k) { w[k] = expf(sS[g * MEMROW + k] - mx); sum += w[k]; }
    float inv = 1.0f / sum;
    float o = 0.0f;
    for (int k = 0; k < MEMROW; ++k)
      o += w[k] * Vi[((size_t)b * MEMROW + k) * HH + h * 128 + d];
    att16[((size_t)b * 2 + q) * HH + h * 128 + d] = (_Float16)(o * inv);
  }
}

// ---------------------------------------------------------------------------
// cls = LN_ic(q + attO); then per-head LN -> f16. Block per (b, q) row.
// ---------------------------------------------------------------------------
__global__ __launch_bounds__(256)
void cls_ln_kernel(const float* __restrict__ attO,
                   const float* __restrict__ lat_cls, const float* __restrict__ lon_cls,
                   const float* __restrict__ icg, const float* __restrict__ icb,
                   const float* __restrict__ latg, const float* __restrict__ latb,
                   const float* __restrict__ long_, const float* __restrict__ lonb,
                   _Float16* __restrict__ hlat16, _Float16* __restrict__ hlon16)
{
  __shared__ float red[256];
  int r = blockIdx.x;
  int t = threadIdx.x;
  int q = r & 1;
  int b = r >> 1;
  const float* qc = q ? lon_cls : lat_cls;

  float v[2]; float s = 0.0f;
  #pragma unroll
  for (int i = 0; i < 2; ++i) {
    int c = t + (i << 8);
    v[i] = qc[c] + attO[(size_t)r * HH + c];
    s += v[i];
  }
  red[t] = s; __syncthreads();
  for (int st = 128; st > 0; st >>= 1) { if (t < st) red[t] += red[t + st]; __syncthreads(); }
  float mean1 = red[0] / (float)HH;
  __syncthreads();
  float s2 = 0.0f;
  #pragma unroll
  for (int i = 0; i < 2; ++i) { float dd = v[i] - mean1; s2 += dd * dd; }
  red[t] = s2; __syncthreads();
  for (int st = 128; st > 0; st >>= 1) { if (t < st) red[t] += red[t + st]; __syncthreads(); }
  float rstd1 = rsqrtf(red[0] / (float)HH + 1e-5f);
  __syncthreads();

  float u[2]; float s3 = 0.0f;
  #pragma unroll
  for (int i = 0; i < 2; ++i) {
    int c = t + (i << 8);
    u[i] = (v[i] - mean1) * rstd1 * icg[c] + icb[c];
    s3 += u[i];
  }
  red[t] = s3; __syncthreads();
  for (int st = 128; st > 0; st >>= 1) { if (t < st) red[t] += red[t + st]; __syncthreads(); }
  float mean2 = red[0] / (float)HH;
  __syncthreads();
  float s4 = 0.0f;
  #pragma unroll
  for (int i = 0; i < 2; ++i) { float dd = u[i] - mean2; s4 += dd * dd; }
  red[t] = s4; __syncthreads();
  for (int st = 128; st > 0; st >>= 1) { if (t < st) red[t] += red[t + st]; __syncthreads(); }
  float rstd2 = rsqrtf(red[0] / (float)HH + 1e-5f);

  const float* hg = q ? long_ : latg;
  const float* hb = q ? lonb  : latb;
  _Float16* outp  = q ? hlon16 : hlat16;
  #pragma unroll
  for (int i = 0; i < 2; ++i) {
    int c = t + (i << 8);
    outp[(size_t)b * HH + c] = (_Float16)((u[i] - mean2) * rstd2 * hg[c] + hb[c]);
  }
}

// ---------------------------------------------------------------------------
// Final head projections (512 -> 3), both heads.
// ---------------------------------------------------------------------------
__global__ __launch_bounds__(256)
void head_out_kernel(const _Float16* __restrict__ zlat, const _Float16* __restrict__ zlon,
                     const float* __restrict__ w2lat, const float* __restrict__ b2lat,
                     const float* __restrict__ w2lon, const float* __restrict__ b2lon,
                     float* __restrict__ outLat, float* __restrict__ outLon)
{
  int i = blockIdx.x * 256 + threadIdx.x;
  if (i >= BB * 6) return;
  int b = i / 6, r = i % 6;
  int head = r / 3, j = r % 3;
  const _Float16* z = (head ? zlon : zlat) + (size_t)b * HH;
  const float* w2 = head ? w2lon : w2lat;
  const float* b2 = head ? b2lon : b2lat;
  float a = b2[j];
  for (int d = 0; d < HH; ++d) a += (float)z[d] * w2[d * 3 + j];
  (head ? outLon : outLat)[b * 3 + j] = a;
}

// ===========================================================================
// Host launcher
// ===========================================================================
extern "C" void kernel_launch(void* const* d_in, const int* in_sizes, int n_in,
                              void* d_out, int out_size, void* d_ws, size_t ws_size,
                              hipStream_t stream)
{
  // -------- inputs (setup_inputs order) --------
  const float* ego_state_norm = (const float*)d_in[0];
  const unsigned char* mask   = (const unsigned char*)d_in[2];
  const float* ego_lane       = (const float*)d_in[3];
  const float* nbr_lane       = (const float*)d_in[4];
  const float* nbr_dist       = (const float*)d_in[5];
  const float* ego_state_raw  = (const float*)d_in[6];
  const float* nbr_state_raw  = (const float*)d_in[7];

  // -------- params (jax pytree flatten: sorted dict keys, DFS) --------
  #define P(i) ((const float*)d_in[(i)])
  const float* eh_b    = P(8);
  const float* eh_ln_b = P(9);
  const float* eh_ln_g = P(10);
  const float* eh_w    = P(11);
  const float* emb_b   = P(12);
  const float* emb_w   = P(13);
  // enc layers: base 14 + 16*L :
  //  +0 bk +1 bo +2 bq +3 bv +4 wk +5 wo +6 wq +7 wv
  //  +8 b1 +9 b2 +10 ln1_b +11 ln1_g +12 ln2_b +13 ln2_g +14 w1 +15 w2
  const float* ic_ln_b = P(62);
  const float* ic_ln_g = P(63);
  const float* im_ln_b = P(64);
  const float* im_ln_g = P(65);
  const float* i_bk = P(66); const float* i_bo = P(67);
  const float* i_bq = P(68); const float* i_bv = P(69);
  const float* i_wk = P(70); const float* i_wo = P(71);
  const float* i_wq = P(72); const float* i_wv = P(73);
  const float* lane_tab = P(74);
  const float* lat_cls  = P(75);
  const float* lat_b1 = P(76); const float* lat_b2 = P(77);
  const float* lat_lnb = P(78); const float* lat_lng = P(79);
  const float* lat_w1 = P(80); const float* lat_w2 = P(81);
  const float* lon_cls  = P(82);
  const float* lon_b1 = P(83); const float* lon_b2 = P(84);
  const float* lon_lnb = P(85); const float* lon_lng = P(86);
  const float* lon_w1 = P(87); const float* lon_w2 = P(88);
  const float* meta_b    = P(89);
  const float* meta_ln_b = P(90);
  const float* meta_ln_g = P(91);
  const float* meta_w    = P(92);
  const float* rel_b1 = P(93); const float* rel_b2 = P(94);
  const float* rel_ln_b = P(95); const float* rel_ln_g = P(96);
  const float* rel_w1 = P(97); const float* rel_w2 = P(98);

  // -------- output regions --------
  float* mem    = (float*)d_out;                               // (B,22,H)
  float* mmask  = mem + (size_t)BB * MEMROW * HH;              // (B,22)
  float* outLat = mmask + (size_t)BB * MEMROW;                 // (B,3)
  float* outLon = outLat + (size_t)BB * 3;                     // (B,3)

  // -------- workspace bump allocator --------
  uintptr_t wsb = (uintptr_t)d_ws;
  size_t off = 0;
  auto alloc = [&](size_t bytes) -> void* {
    void* p = (void*)(wsb + off);
    off = (off + bytes + 255) & ~(size_t)255;
    return p;
  };
  const size_t ELW = (size_t)ROWS * DD;      // 65536*256

  float*    x     = (float*)alloc(ELW * 4);
  float*    pos   = (float*)alloc(ELW * 4);
  _Float16* qk16  = (_Float16*)alloc(ELW * 2);
  _Float16* x16   = (_Float16*)alloc(ELW * 2);
  _Float16* eh16  = (_Float16*)alloc(ELW * 2);
  _Float16* kv16  = (_Float16*)alloc((size_t)BB * MEMROW * HH * 2);
  _Float16* s16r  = (_Float16*)alloc((size_t)BB * KK * HH * 2);
  float*    feat  = (float*)alloc((size_t)BB * KK * 9 * 4);
  float*    selv  = (float*)alloc((size_t)BB * KK * 4);
  float*    Qc    = (float*)alloc(2 * HH * 4);
  _Float16* hlat16 = (_Float16*)alloc((size_t)BB * HH * 2);
  _Float16* hlon16 = (_Float16*)alloc((size_t)BB * HH * 2);
  _Float16* zlat16 = (_Float16*)alloc((size_t)BB * HH * 2);
  _Float16* zlon16 = (_Float16*)alloc((size_t)BB * HH * 2);
  int*      cnt   = (int*)alloc(BB * 4);
  int*      baseA = (int*)alloc(BB * 4);

  // f16 transposed weights (Wt: N x K)
  _Float16 *wq16[3], *wk16[3], *wv16[3], *wo16[3], *w116[3], *w216[3];
  for (int L = 0; L < 3; ++L) {
    wq16[L] = (_Float16*)alloc((size_t)DD * DD * 2);
    wk16[L] = (_Float16*)alloc((size_t)DD * DD * 2);
    wv16[L] = (_Float16*)alloc((size_t)DD * DD * 2);
    wo16[L] = (_Float16*)alloc((size_t)DD * DD * 2);
    w116[L] = (_Float16*)alloc((size_t)DD * FFD * 2);
    w216[L] = (_Float16*)alloc((size_t)FFD * DD * 2);
  }
  _Float16* ehw16  = (_Float16*)alloc((size_t)DD * HH * 2);
  _Float16* relw216 = (_Float16*)alloc((size_t)HH * HH * 2);
  _Float16* iwk16  = (_Float16*)alloc((size_t)HH * HH * 2);
  _Float16* iwv16  = (_Float16*)alloc((size_t)HH * HH * 2);
  _Float16* iwo16  = (_Float16*)alloc((size_t)HH * HH * 2);
  _Float16* latw116 = (_Float16*)alloc((size_t)HH * HH * 2);
  _Float16* lonw116 = (_Float16*)alloc((size_t)HH * HH * 2);

  // union scratch region: encoder phase vs intent phase
  size_t uoff = off;
  auto ualloc = [&](size_t& o, size_t bytes) -> void* {
    void* p = (void*)(wsb + o);
    o = (o + bytes + 255) & ~(size_t)255;
    return p;
  };
  size_t oe = uoff;
  float*    Qb     = (float*)ualloc(oe, ELW * 4);
  float*    Kb     = (float*)ualloc(oe, ELW * 4);
  float*    Vb     = (float*)ualloc(oe, ELW * 4);
  float*    attn32 = (float*)ualloc(oe, ELW * 4);
  _Float16* attn16 = (_Float16*)ualloc(oe, ELW * 2);
  float*    proj   = (float*)ualloc(oe, ELW * 4);
  _Float16* h116   = (_Float16*)ualloc(oe, (size_t)ROWS * FFD * 2);
  size_t oi = uoff;
  float*    Ki     = (float*)ualloc(oi, (size_t)BB * MEMROW * HH * 4);
  float*    Vi     = (float*)ualloc(oi, (size_t)BB * MEMROW * HH * 4);
  _Float16* att16i = (_Float16*)ualloc(oi, (size_t)BB * 2 * HH * 2);
  float*    attO   = (float*)ualloc(oi, (size_t)BB * 2 * HH * 4);

  // -------- helpers --------
  auto cvt_t = [&](const float* s, _Float16* d, int Kd, int Nd) {
    f2h_t_kernel<<<(Kd * Nd + 255) / 256, 256, 0, stream>>>(s, d, Kd, Nd);
  };
  auto gemm = [&](int ACT, const _Float16* A, const _Float16* Wt, const float* bias,
                  float* C32, _Float16* C16, const float* scale,
                  int M, int N, int K, int rdiv, int rmul, int radd) {
    dim3 g(M / 128, N / 64), blk(256);
    switch (ACT) {
      case 0: wmma_gemm<0><<<g, blk, 0, stream>>>(A, Wt, bias, C32, C16, scale, M, N, K, rdiv, rmul, radd); break;
      case 1: wmma_gemm<1><<<g, blk, 0, stream>>>(A, Wt, bias, C32, C16, scale, M, N, K, rdiv, rmul, radd); break;
      case 2: wmma_gemm<2><<<g, blk, 0, stream>>>(A, Wt, bias, C32, C16, scale, M, N, K, rdiv, rmul, radd); break;
      default: wmma_gemm<3><<<g, blk, 0, stream>>>(A, Wt, bias, C32, C16, scale, M, N, K, rdiv, rmul, radd); break;
    }
  };

  // -------- weight conversion (transposed to N x K) --------
  for (int L = 0; L < 3; ++L) {
    int b = 14 + 16 * L;
    cvt_t(P(b + 6), wq16[L], DD, DD);
    cvt_t(P(b + 4), wk16[L], DD, DD);
    cvt_t(P(b + 7), wv16[L], DD, DD);
    cvt_t(P(b + 5), wo16[L], DD, DD);
    cvt_t(P(b + 14), w116[L], DD, FFD);
    cvt_t(P(b + 15), w216[L], FFD, DD);
  }
  cvt_t(eh_w, ehw16, DD, HH);
  cvt_t(rel_w2, relw216, HH, HH);
  cvt_t(i_wk, iwk16, HH, HH);
  cvt_t(i_wv, iwv16, HH, HH);
  cvt_t(i_wo, iwo16, HH, HH);
  cvt_t(lat_w1, latw116, HH, HH);
  cvt_t(lon_w1, lonw116, HH, HH);

  // -------- embedding + positional encoding --------
  embed_pos_kernel<<<ROWS, 256, 0, stream>>>(ego_state_norm, emb_w, emb_b, x, pos, x16, qk16);

  // -------- encoder layers --------
  for (int L = 0; L < 3; ++L) {
    int b = 14 + 16 * L;
    gemm(0, qk16, wq16[L], P(b + 2), Qb, nullptr, nullptr, ROWS, DD, DD, 0, 0, 0);
    gemm(0, qk16, wk16[L], P(b + 0), Kb, nullptr, nullptr, ROWS, DD, DD, 0, 0, 0);
    gemm(0, x16,  wv16[L], P(b + 3), Vb, nullptr, nullptr, ROWS, DD, DD, 0, 0, 0);
    attn_enc_kernel<<<BB, 128, 0, stream>>>(Qb, Kb, Vb, attn32, attn16);
    gemm(0, attn16, wo16[L], P(b + 1), proj, nullptr, nullptr, ROWS, DD, DD, 0, 0, 0);
    ln_kernel<<<ROWS, 256, 0, stream>>>(x, proj, P(b + 11), P(b + 10), x, x16,
                                        nullptr, nullptr, DD);
    gemm(1, x16, w116[L], P(b + 8), nullptr, h116, nullptr, ROWS, FFD, DD, 0, 0, 0);
    gemm(0, h116, w216[L], P(b + 9), proj, nullptr, nullptr, ROWS, DD, FFD, 0, 0, 0);
    ln_kernel<<<ROWS, 256, 0, stream>>>(x, proj, P(b + 13), P(b + 12), x, x16,
                                        pos, qk16, DD);
  }

  // -------- meta + ego_hidden --------
  meta_kernel<<<ROWS, 256, 0, stream>>>(x, ego_lane, lane_tab, meta_ln_g, meta_ln_b,
                                        meta_w, meta_b, eh_ln_g, eh_ln_b, eh16);
  gemm(2, eh16, ehw16, eh_b, mem, nullptr, nullptr, ROWS, HH, DD, TT, MEMROW, 0);

  // -------- neighbor selection + rel tokens --------
  cnt_kernel<<<BB / 256, 256, 0, stream>>>(mask, cnt);
  scan_kernel<<<1, 256, 0, stream>>>(cnt, baseA);
  select_kernel<<<BB, 32, 0, stream>>>(mask, baseA, nbr_state_raw, nbr_lane, nbr_dist,
                                       ego_state_raw, ego_lane, feat, selv, mmask);
  rel_mlp1_kernel<<<BB * KK, 256, 0, stream>>>(feat, rel_ln_g, rel_ln_b, rel_w1, rel_b1, s16r);
  gemm(0, s16r, relw216, rel_b2, mem, nullptr, selv, BB * KK, HH, HH, KK, MEMROW, TT);

  // -------- intent attention --------
  ln_kernel<<<BB * MEMROW, 256, 0, stream>>>(mem, nullptr, im_ln_g, im_ln_b,
                                             nullptr, kv16, nullptr, nullptr, HH);
  qc_kernel<<<2, HH, 0, stream>>>(lat_cls, lon_cls, i_wq, i_bq, Qc);
  gemm(0, kv16, iwk16, i_bk, Ki, nullptr, nullptr, BB * MEMROW, HH, HH, 0, 0, 0);
  gemm(0, kv16, iwv16, i_bv, Vi, nullptr, nullptr, BB * MEMROW, HH, HH, 0, 0, 0);
  intent_attn_kernel<<<BB, 256, 0, stream>>>(Qc, Ki, Vi, mmask, att16i);
  gemm(0, att16i, iwo16, i_bo, attO, nullptr, nullptr, BB * 2, HH, HH, 0, 0, 0);
  cls_ln_kernel<<<BB * 2, 256, 0, stream>>>(attO, lat_cls, lon_cls, ic_ln_g, ic_ln_b,
                                            lat_lng, lat_lnb, lon_lng, lon_lnb,
                                            hlat16, hlon16);
  gemm(3, hlat16, latw116, lat_b1, nullptr, zlat16, nullptr, BB, HH, HH, 0, 0, 0);
  gemm(3, hlon16, lonw116, lon_b1, nullptr, zlon16, nullptr, BB, HH, HH, 0, 0, 0);
  head_out_kernel<<<(BB * 6 + 255) / 256, 256, 0, stream>>>(zlat16, zlon16,
                                                            lat_w2, lat_b2, lon_w2, lon_b2,
                                                            outLat, outLon);
  (void)in_sizes; (void)n_in; (void)out_size; (void)ws_size;
}
